// Attention_88656714924472
// MI455X (gfx1250) — compile-verified, matrix operand order checked by reading, not run on hardware
//
#include <hip/hip_runtime.h>
#include <hip/hip_bf16.h>
#include <math.h>

// Problem constants (B=16, N=1024, D=768, H=12, HD=64, NUM_BUCKETS=49)
#define Bc   16
#define Nc_  1024
#define Dc   768
#define Hc   12
#define HDc  64
#define BHc  (Bc * Hc)        // 192
#define MTOK (Bc * Nc_)       // 16384 tokens
#define MQ   (BHc * Nc_)      // 196608 (bh,n) rows

typedef __attribute__((ext_vector_type(16))) __bf16 v16bf;
typedef __attribute__((ext_vector_type(8)))  float  v8f;

__device__ __forceinline__ v16bf load_frag(const __bf16* p) {
  // 16 bf16 per lane = two contiguous 16B chunks (K and K+16 in the wave32 layout)
  union { v16bf v; uint4 q[2]; } f;
  f.q[0] = *(const uint4*)p;
  f.q[1] = *(const uint4*)(p + 16);
  return f.v;
}

__device__ __forceinline__ v8f wmma_bf16(v16bf a, v16bf b, v8f c) {
  return __builtin_amdgcn_wmma_f32_16x16x32_bf16(false, a, false, b, (short)0, c,
                                                 false, false);
}

// ---- CDNA5 async global->LDS copy (ASYNCcnt path, no VGPR round-trip) ------
__device__ __forceinline__ void async_copy_b128(void* lds_dst, const void* gsrc) {
  unsigned lds_off = (unsigned)(size_t)lds_dst;  // addrspace(3) offset
  asm volatile("global_load_async_to_lds_b128 %0, %1, off"
               :: "v"(lds_off), "v"(gsrc)
               : "memory");
}
__device__ __forceinline__ void wait_async0() {
  asm volatile("s_wait_asynccnt 0x0" ::: "memory");
}

// ---------------------------------------------------------------- conversions
__global__ void cvt_f32_bf16(const float* __restrict__ s, __bf16* __restrict__ d,
                             size_t n) {
  size_t i = (size_t)blockIdx.x * blockDim.x + threadIdx.x;
  if (i < n) d[i] = (__bf16)s[i];
}

// rpe_table (49,64) -> padded (64,64) bf16, zeros for buckets >= 49
__global__ void pad_rpe(const float* __restrict__ s, __bf16* __restrict__ d) {
  int c = blockIdx.x;   // 0..63
  int k = threadIdx.x;  // 0..63
  d[c * HDc + k] = (c < 49) ? (__bf16)s[c * HDc + k] : (__bf16)0.0f;
}

// ------------------------------------------------------- generic bf16 GEMM
// C[M,Nc] = A[M,K] * W[Nc,K]^T  (both row-major bf16), f32 accum, epilogue cb.
// WG = 256 threads = 8 waves; 128x128 output tile; wave = 32 rows x 64 cols.
template <class Epi>
__global__ __launch_bounds__(256)
void gemm_bf16_tn(const __bf16* __restrict__ A, const __bf16* __restrict__ W,
                  int K, Epi epi) {
  __shared__ __bf16 As[128 * 32];
  __shared__ __bf16 Ws[128 * 32];
  const int wave = threadIdx.x >> 5, lane = threadIdx.x & 31;
  const int lrow = lane & 15, hi = lane >> 4;
  const int wr = wave & 3;   // 4 row-groups of 32
  const int wc = wave >> 2;  // 2 col-groups of 64
  const int m0 = blockIdx.y * 128;
  const int c0 = blockIdx.x * 128;
  const int tr = threadIdx.x >> 1, tk = (threadIdx.x & 1) * 16;

  v8f acc[2][4] = {};

  for (int k0 = 0; k0 < K; k0 += 32) {
    __syncthreads();
    // async global->LDS tile staging (16B per lane per tile)
    async_copy_b128(&As[tr * 32 + tk], &A[(size_t)(m0 + tr) * K + k0 + tk]);
    async_copy_b128(&Ws[tr * 32 + tk], &W[(size_t)(c0 + tr) * K + k0 + tk]);
    __builtin_prefetch(&A[(size_t)(m0 + tr) * K + k0 + 32 + tk], 0, 1);
    __builtin_prefetch(&W[(size_t)(c0 + tr) * K + k0 + 32 + tk], 0, 1);
    wait_async0();
    __syncthreads();

    v16bf af[2], wf[4];
#pragma unroll
    for (int i = 0; i < 2; i++)
      af[i] = load_frag(&As[(wr * 32 + i * 16 + lrow) * 32 + hi * 8]);
#pragma unroll
    for (int j = 0; j < 4; j++)
      wf[j] = load_frag(&Ws[(wc * 64 + j * 16 + lrow) * 32 + hi * 8]);
#pragma unroll
    for (int i = 0; i < 2; i++)
#pragma unroll
      for (int j = 0; j < 4; j++)
        acc[i][j] = wmma_bf16(af[i], wf[j], acc[i][j]);
  }

#pragma unroll
  for (int i = 0; i < 2; i++)
#pragma unroll
    for (int j = 0; j < 4; j++)
#pragma unroll
      for (int r = 0; r < 8; r++) {
        int m = m0 + wr * 32 + i * 16 + r + hi * 8;
        int c = c0 + wc * 64 + j * 16 + lrow;
        epi(m, c, acc[i][j][r]);
      }
}

// QKV epilogue: scatter into Q (pre-scaled, bh-major), K (bh-major), V^T (HD-major)
struct QKVEpi {
  __bf16 *Q, *K, *VT;
  __device__ void operator()(int m, int c, float v) const {
    int b = m >> 10, n = m & 1023;
    int s = c / Dc, rem = c - s * Dc;
    int h = rem >> 6, d = rem & 63;
    size_t bh = (size_t)(b * Hc + h);
    if (s == 0)      Q [(bh * Nc_ + n) * HDc + d] = (__bf16)(v * 0.125f);
    else if (s == 1) K [(bh * Nc_ + n) * HDc + d] = (__bf16)v;
    else             VT[(bh * HDc + d) * Nc_ + n] = (__bf16)v;
  }
};

struct ProjEpi {
  const float* bias;
  float* out;
  __device__ void operator()(int m, int c, float v) const {
    out[(size_t)m * Dc + c] = v + bias[c];
  }
};

// ------------------------------------------------- bias_all = Q * rpe_pad^T
// A rows = (bh*N + n), K=64; output (MQ, 64) f32.  One wave = 16 rows x 64 cols.
__global__ __launch_bounds__(256)
void rpe_bias_gemm(const __bf16* __restrict__ Q, const __bf16* __restrict__ R,
                   float* __restrict__ biasAll) {
  const int wave = threadIdx.x >> 5, lane = threadIdx.x & 31;
  const int lrow = lane & 15, hi = lane >> 4;
  const size_t m0 = (size_t)blockIdx.x * 128 + wave * 16;

  v16bf a[2];
#pragma unroll
  for (int kk = 0; kk < 2; kk++)
    a[kk] = load_frag(Q + (m0 + lrow) * HDc + kk * 32 + hi * 8);

  v8f acc[4] = {};
#pragma unroll
  for (int kk = 0; kk < 2; kk++)
#pragma unroll
    for (int j = 0; j < 4; j++) {
      v16bf bfr = load_frag(R + (j * 16 + lrow) * HDc + kk * 32 + hi * 8);
      acc[j] = wmma_bf16(a[kk], bfr, acc[j]);
    }
#pragma unroll
  for (int j = 0; j < 4; j++)
#pragma unroll
    for (int r = 0; r < 8; r++)
      biasAll[(m0 + r + hi * 8) * 64 + j * 16 + lrow] = acc[j][r];
}

// ------------------------------------------------------------ flash attention
// grid = (N/128, B*H); WG = 8 waves; wave owns 16 q rows (q in registers),
// streams 64-wide K / V^T tiles through LDS with online softmax.
__global__ __launch_bounds__(256)
void flash_attn(const __bf16* __restrict__ Q, const __bf16* __restrict__ Kt,
                const __bf16* __restrict__ VT, const float* __restrict__ biasAll,
                const int* __restrict__ rp, __bf16* __restrict__ AOut) {
  __shared__ __bf16 Ks[64 * 64];
  __shared__ __bf16 Vs[64 * 64];       // Vs[d][j] (V transposed)
  __shared__ __bf16 Ps[8][16 * 32];    // per-wave P transpose buffer

  const int wave = threadIdx.x >> 5, lane = threadIdx.x & 31;
  const int lrow = lane & 15, hi = lane >> 4;
  const int bh = blockIdx.y;
  const int b = bh / Hc, h = bh - b * Hc;
  const int i0 = blockIdx.x * 128 + wave * 16;

  const __bf16* Qb = Q  + (size_t)bh * Nc_ * HDc;
  const __bf16* Kb = Kt + (size_t)bh * Nc_ * HDc;
  const __bf16* Vb = VT + (size_t)bh * HDc * Nc_;
  const float*  Bb = biasAll + (size_t)bh * Nc_ * 64;

  v16bf qa[2];
  qa[0] = load_frag(Qb + (size_t)(i0 + lrow) * HDc + 0  + hi * 8);
  qa[1] = load_frag(Qb + (size_t)(i0 + lrow) * HDc + 32 + hi * 8);

  v8f o[4] = {};
  float mi[8], li[8];
#pragma unroll
  for (int r = 0; r < 8; r++) { mi[r] = -INFINITY; li[r] = 0.0f; }

  for (int j0 = 0; j0 < Nc_; j0 += 64) {
    __syncthreads();
    {
      // async global->LDS staging of K tile and V^T tile (16B per lane each)
      int r = threadIdx.x >> 2, cc = (threadIdx.x & 3) * 16;
      async_copy_b128(&Ks[r * 64 + cc], &Kb[(size_t)(j0 + r) * HDc + cc]);
      async_copy_b128(&Vs[r * 64 + cc], &Vb[(size_t)r * Nc_ + j0 + cc]);
      wait_async0();
    }
    __syncthreads();

#pragma unroll
    for (int js = 0; js < 64; js += 32) {
      // S tile: 16 rows x 32 cols = two C fragments, K = HD = 64
      v8f s0 = {}, s1 = {};
#pragma unroll
      for (int kk = 0; kk < 2; kk++) {
        v16bf b0 = load_frag(&Ks[(js + lrow) * 64 + kk * 32 + hi * 8]);
        v16bf b1 = load_frag(&Ks[(js + 16 + lrow) * 64 + kk * 32 + hi * 8]);
        s0 = wmma_bf16(qa[kk], b0, s0);
        s1 = wmma_bf16(qa[kk], b1, s1);
      }

      // relative-position bias gather
      const int jc0 = j0 + js + lrow;
      float sv0[8], sv1[8];
#pragma unroll
      for (int r = 0; r < 8; r++) {
        int irow = i0 + r + hi * 8;
        int bk0 = rp[(size_t)irow * Nc_ + jc0];
        int bk1 = rp[(size_t)irow * Nc_ + jc0 + 16];
        sv0[r] = s0[r] + Bb[(size_t)irow * 64 + bk0];
        sv1[r] = s1[r] + Bb[(size_t)irow * 64 + bk1];
      }

      // online softmax (row = VGPR idx + 8*hi; 16 lanes per row-group)
#pragma unroll
      for (int r = 0; r < 8; r++) {
        float mx = fmaxf(sv0[r], sv1[r]);
#pragma unroll
        for (int off = 1; off < 16; off <<= 1)
          mx = fmaxf(mx, __shfl_xor(mx, off, 32));
        float mn = fmaxf(mi[r], mx);
        float sc = __expf(mi[r] - mn);
        float p0 = __expf(sv0[r] - mn);
        float p1 = __expf(sv1[r] - mn);
        float ps = p0 + p1;
#pragma unroll
        for (int off = 1; off < 16; off <<= 1)
          ps += __shfl_xor(ps, off, 32);
        li[r] = li[r] * sc + ps;
        mi[r] = mn;
        o[0][r] *= sc; o[1][r] *= sc; o[2][r] *= sc; o[3][r] *= sc;
        int ii = r + hi * 8;
        Ps[wave][ii * 32 + lrow]      = (__bf16)p0;
        Ps[wave][ii * 32 + lrow + 16] = (__bf16)p1;
      }
      __syncthreads();  // uniform across WG; orders C->A transpose through LDS

      // O += P * V  (A = P 16x32, B = V^T rows from Vs)
      v16bf pa = load_frag(&Ps[wave][lrow * 32 + hi * 8]);
#pragma unroll
      for (int n = 0; n < 4; n++) {
        v16bf vb = load_frag(&Vs[(n * 16 + lrow) * 64 + js + hi * 8]);
        o[n] = wmma_bf16(pa, vb, o[n]);
      }
    }
  }

  // normalize and store bf16 (b, n, h*64+d) for the projection GEMM
#pragma unroll
  for (int r = 0; r < 8; r++) {
    float inv = 1.0f / li[r];
    int irow = i0 + r + hi * 8;
#pragma unroll
    for (int n = 0; n < 4; n++) {
      AOut[((size_t)(b * Nc_ + irow)) * Dc + h * HDc + n * 16 + lrow] =
          (__bf16)(o[n][r] * inv);
    }
  }
}

// ---------------------------------------------------------------- launcher
extern "C" void kernel_launch(void* const* d_in, const int* in_sizes, int n_in,
                              void* d_out, int out_size, void* d_ws, size_t ws_size,
                              hipStream_t stream) {
  const float* x      = (const float*)d_in[0];
  const float* qkv_w  = (const float*)d_in[1];
  const float* rpe    = (const float*)d_in[2];
  const int*   rp     = (const int*)d_in[3];
  const float* proj_w = (const float*)d_in[4];
  const float* proj_b = (const float*)d_in[5];
  float* out = (float*)d_out;

  char* ws = (char*)d_ws;
  size_t off = 0;
  auto alloc = [&](size_t bytes) -> void* {
    void* p = ws + off;
    off = (off + bytes + 255) & ~(size_t)255;
    return p;
  };

  __bf16* Xb    = (__bf16*)alloc((size_t)MTOK * Dc * 2);
  __bf16* Wqkv  = (__bf16*)alloc((size_t)3 * Dc * Dc * 2);
  __bf16* Wproj = (__bf16*)alloc((size_t)Dc * Dc * 2);
  __bf16* Rpad  = (__bf16*)alloc((size_t)64 * 64 * 2);
  __bf16* Qb    = (__bf16*)alloc((size_t)MQ * HDc * 2);
  __bf16* Kb    = (__bf16*)alloc((size_t)MQ * HDc * 2);
  __bf16* VTb   = (__bf16*)alloc((size_t)MQ * HDc * 2);
  float*  biasA = (float*) alloc((size_t)MQ * 64 * 4);
  __bf16* AOut  = (__bf16*)alloc((size_t)MTOK * Dc * 2);

  // 1) precision conversion
  {
    size_t n = (size_t)MTOK * Dc;
    cvt_f32_bf16<<<(unsigned)((n + 255) / 256), 256, 0, stream>>>(x, Xb, n);
    n = (size_t)3 * Dc * Dc;
    cvt_f32_bf16<<<(unsigned)((n + 255) / 256), 256, 0, stream>>>(qkv_w, Wqkv, n);
    n = (size_t)Dc * Dc;
    cvt_f32_bf16<<<(unsigned)((n + 255) / 256), 256, 0, stream>>>(proj_w, Wproj, n);
    pad_rpe<<<64, 64, 0, stream>>>(rpe, Rpad);
  }

  // 2) QKV GEMM: (16384 x 768) * (2304 x 768)^T -> scatter Q(scaled)/K/V^T
  gemm_bf16_tn<<<dim3((3 * Dc) / 128, MTOK / 128), 256, 0, stream>>>(
      Xb, Wqkv, Dc, QKVEpi{Qb, Kb, VTb});

  // 3) bias_all = Q * rpe_pad^T  (MQ x 64)
  rpe_bias_gemm<<<MQ / 128, 256, 0, stream>>>(Qb, Rpad, biasA);

  // 4) flash attention with gathered rpe bias
  flash_attn<<<dim3(Nc_ / 128, BHc), 256, 0, stream>>>(Qb, Kb, VTb, biasA, rp,
                                                       AOut);

  // 5) output projection: (16384 x 768) * (768 x 768)^T + b
  gemm_bf16_tn<<<dim3(Dc / 128, MTOK / 128), 256, 0, stream>>>(
      AOut, Wproj, Dc, ProjEpi{proj_b, out});
}
